// get_model_75969381532361
// MI455X (gfx1250) — matrix operand order repaired and checked by reading, not compiled, not run on hardware
//
#include <hip/hip_runtime.h>
#include <hip/hip_bf16.h>

typedef _Float16 hf16;
typedef __attribute__((ext_vector_type(16))) _Float16 v16h;
typedef __attribute__((ext_vector_type(4)))  _Float16 v4h;
typedef __attribute__((ext_vector_type(8)))  float    v8f;

#define BN_EPS 1e-5f

// ---------------------------------------------------------------------------
// WMMA fragment helpers (16x16x32 f16 -> f32), wave32 layouts per CDNA5 ISA:
//  A (16x32 f16): lane m=lane&15, grp=lane>>4; elem j -> k = (j<8? j : j+8)+8*grp
//    => per lane two contiguous 8-half runs -> 2x ds_load_b128
//  B (32x16 f16): lane n=lane&15, grp=lane>>4; elem j -> k = (j<8? j : j+8)+8*grp
//    => weights PRE-SWIZZLED fragment-major: [nt][kt][lane][16] so a B-frag is
//       one contiguous v16h (32B) per lane -> 2x global_load_b128
//  C/D (16x16 f32): lane n=lane&15; vgpr r -> m = r + 8*(lane>>4)
// ---------------------------------------------------------------------------
__device__ __forceinline__ v16h frag_a_lds(const hf16* buf, int stride, int lane, int k0) {
    const int m   = lane & 15;
    const int grp = lane >> 4;
    v16h a;
#pragma unroll
    for (int j = 0; j < 16; ++j) {
        const int k = ((j < 8) ? j : (j + 8)) + (grp << 3);
        a[j] = buf[m * stride + k0 + k];
    }
    return a;
}

__device__ __forceinline__ v16h frag_a_global_f32(const float* x, int stride, int lane,
                                                  int k0, int m0) {
    const int m   = m0 + (lane & 15);
    const int grp = lane >> 4;
    v16h a;
#pragma unroll
    for (int j = 0; j < 16; ++j) {
        const int k = ((j < 8) ? j : (j + 8)) + (grp << 3);
        a[j] = (hf16)x[m * stride + k0 + k];
    }
    return a;
}

// Swizzled B fragment: weights stored as [nt][kt][lane][16 halves] (32B/lane).
__device__ __forceinline__ v16h frag_b_swz(const hf16* w, int KT, int lane, int kt, int nt) {
    const v16h* p = (const v16h*)w;
    return p[(nt * KT + kt) * 32 + lane];
}

// ---------------------------------------------------------------------------
// Fold BN into weights/bias; emit fp16 weights in WMMA-fragment-swizzled
// layout [nt][kt][lane][16] (k = kt*32 + (j<8? j : j+8) + 8*(lane>>4), n = nt*16 + lane&15),
// fp32 bias. cinPad % 32 == 0, coutPad % 16 == 0.
// cfPerm > 0: input channels are permuted [features(cfPerm) | xyz(3)] so weight
// column k maps to original column (k < cfPerm ? k+3 : k-cfPerm).
// ---------------------------------------------------------------------------
__global__ void fold_kernel(const float* __restrict__ W, const float* __restrict__ b,
                            const float* __restrict__ g, const float* __restrict__ be,
                            const float* __restrict__ m, const float* __restrict__ v,
                            int cin, int cout, int cinPad, int coutPad,
                            hf16* __restrict__ Wout, float* __restrict__ bOut,
                            int has_bn, int cfPerm) {
    const int t = blockIdx.x * blockDim.x + threadIdx.x;
    const int total = cinPad * coutPad;
    if (t < total) {
        const int j    = t & 15;
        const int lane = (t >> 4) & 31;
        const int blk  = t >> 9;
        const int KT   = cinPad >> 5;
        const int kt   = blk % KT;
        const int nt   = blk / KT;
        const int n = (nt << 4) + (lane & 15);
        const int k = (kt << 5) + ((j < 8) ? j : (j + 8)) + ((lane >> 4) << 3);
        float val = 0.f;
        if (k < cin && n < cout) {
            const int korig = (cfPerm > 0) ? ((k < cfPerm) ? k + 3 : k - cfPerm) : k;
            const float scale = has_bn ? g[n] * rsqrtf(v[n] + BN_EPS) : 1.f;
            val = W[n * cin + korig] * scale;
        }
        Wout[t] = (hf16)val;
    }
    if (t < coutPad) {
        float bb = 0.f;
        if (t < cout) {
            if (has_bn) {
                const float scale = g[t] * rsqrtf(v[t] + BN_EPS);
                bb = (b[t] - m[t]) * scale + be[t];
            } else {
                bb = b[t];
            }
        }
        bOut[t] = bb;
    }
}

// ---------------------------------------------------------------------------
// Transpose (B,3,N) -> (B,N,3)
// ---------------------------------------------------------------------------
__global__ void transpose_xyz_kernel(const float* __restrict__ in, float* __restrict__ out,
                                     int B, int N) {
    const int t = blockIdx.x * blockDim.x + threadIdx.x;
    if (t >= B * N * 3) return;
    const int b = t / (N * 3);
    const int r = t - b * N * 3;
    const int n = r / 3;
    const int c = r - n * 3;
    out[t] = in[(b * 3 + c) * N + n];
}

__global__ void zero_kernel(float* __restrict__ p, int n) {
    const int t = blockIdx.x * blockDim.x + threadIdx.x;
    if (t < n) p[t] = 0.f;
}

// ---------------------------------------------------------------------------
// Farthest point sampling: one block per batch, points resident in LDS.
// ---------------------------------------------------------------------------
__global__ void __launch_bounds__(256) fps_kernel(const float* __restrict__ pts,
                                                  int N, int npoint, int* __restrict__ fidx) {
    __shared__ float sx[1024], sy[1024], sz[1024], sdist[1024];
    __shared__ float rv[256];
    __shared__ int   ri[256];
    __shared__ int   s_sel;
    const int b   = blockIdx.x;
    const int tid = threadIdx.x;

    for (int p = tid; p < N; p += 256) {
        sx[p] = pts[(b * N + p) * 3 + 0];
        sy[p] = pts[(b * N + p) * 3 + 1];
        sz[p] = pts[(b * N + p) * 3 + 2];
        sdist[p] = 1e10f;
    }
    __syncthreads();

    int sel = 0;
    for (int i = 0; i < npoint; ++i) {
        if (tid == 0) fidx[b * npoint + i] = sel;
        const float cx = sx[sel], cy = sy[sel], cz = sz[sel];
        float bv = -1.f;
        int   bi = 0;
        for (int p = tid; p < N; p += 256) {
            const float dx = sx[p] - cx, dy = sy[p] - cy, dz = sz[p] - cz;
            const float d  = dx * dx + dy * dy + dz * dz;
            const float nd = fminf(sdist[p], d);
            sdist[p] = nd;
            if (nd > bv) { bv = nd; bi = p; }
        }
        rv[tid] = bv;
        ri[tid] = bi;
        __syncthreads();
        for (int off = 128; off > 0; off >>= 1) {
            if (tid < off) {
                const bool take = (rv[tid + off] > rv[tid]) ||
                                  (rv[tid + off] == rv[tid] && ri[tid + off] < ri[tid]);
                if (take) { rv[tid] = rv[tid + off]; ri[tid] = ri[tid + off]; }
            }
            __syncthreads();
        }
        if (tid == 0) s_sel = ri[0];
        __syncthreads();
        sel = s_sel;
        __syncthreads();
    }
}

// ---------------------------------------------------------------------------
// Gather 3D coords by index: out[b,s,:] = in[b, idx[b,s], :]
// ---------------------------------------------------------------------------
__global__ void gather3_kernel(const float* __restrict__ in, const int* __restrict__ idx,
                               int Np, int S, float* __restrict__ out) {
    const int t = blockIdx.x * blockDim.x + threadIdx.x;
    const int b = t / (S * 3);
    const int r = t - b * S * 3;
    const int s = r / 3;
    const int c = r - s * 3;
    out[t] = in[(b * Np + idx[b * S + s]) * 3 + c];
}

// ---------------------------------------------------------------------------
// Ball query: first K in-radius indices (ascending), padded with first hit.
// ---------------------------------------------------------------------------
__global__ void ball_query_kernel(const float* __restrict__ xyz, const float* __restrict__ nxyz,
                                  int Np, int S, int K, float r2, int B,
                                  int* __restrict__ gidx) {
    const int t = blockIdx.x * blockDim.x + threadIdx.x;
    if (t >= B * S) return;
    const int b = t / S;
    const int s = t - b * S;
    const float cx = nxyz[(b * S + s) * 3 + 0];
    const float cy = nxyz[(b * S + s) * 3 + 1];
    const float cz = nxyz[(b * S + s) * 3 + 2];
    const int base = (b * S + s) * K;
    int cnt = 0, first = 0;
    bool haveFirst = false;
    for (int n = 0; n < Np; ++n) {
        const float dx = xyz[(b * Np + n) * 3 + 0] - cx;
        const float dy = xyz[(b * Np + n) * 3 + 1] - cy;
        const float dz = xyz[(b * Np + n) * 3 + 2] - cz;
        const float d = dx * dx + dy * dy + dz * dz;
        if (d <= r2) {
            if (!haveFirst) { first = n; haveFirst = true; }
            if (cnt < K) gidx[base + cnt++] = n;
        }
    }
    for (; cnt < K; ++cnt) gidx[base + cnt] = first;
}

// ---------------------------------------------------------------------------
// One MLP layer on a 16-sample chunk: WMMA f16->f32, +bias, ReLU.
// LAST layer max-pools into `pooled` (non-negative float-as-int atomicMax),
// else stores f16 activations to `nxt` with row stride COUT.
// All tile loops wave-uniform => EXEC all-ones at every WMMA.
// ---------------------------------------------------------------------------
template <int CIN, int COUT, bool LAST>
__device__ __forceinline__ void mlp_layer(const hf16* cur, hf16* nxt,
                                          const hf16* __restrict__ w,
                                          const float* __restrict__ bias,
                                          float* pooled, int lane, int wave) {
    constexpr int KT = CIN >> 5;
    constexpr int NT = COUT >> 4;
    for (int nt = wave; nt < NT; nt += 4) {
        v8f acc = {};
#pragma unroll
        for (int kt = 0; kt < KT; ++kt) {
            v16h a  = frag_a_lds(cur, CIN, lane, kt << 5);
            v16h bb = frag_b_swz(w, KT, lane, kt, nt);
            acc = __builtin_amdgcn_wmma_f32_16x16x32_f16(
                false, a, false, bb, (short)0, acc, false, false);
        }
        const int n    = lane & 15;
        const int grp  = lane >> 4;
        const int ncol = (nt << 4) + n;
        const float bs = bias[ncol];
        if constexpr (LAST) {
            float mx = 0.f;  // ReLU floor
#pragma unroll
            for (int j = 0; j < 8; ++j) mx = fmaxf(mx, acc[j] + bs);
            atomicMax((int*)&pooled[ncol], __float_as_int(mx));
        } else {
#pragma unroll
            for (int j = 0; j < 8; ++j) {
                const float vv = fmaxf(acc[j] + bs, 0.f);
                nxt[(j + (grp << 3)) * COUT + ncol] = (hf16)vv;
            }
        }
    }
}

// ---------------------------------------------------------------------------
// Fused gather -> 3-layer MLP -> ReLU -> max-pool over group.
// Blocks: (s, b, chunk-block); 128 threads = 4 waves. All dims compile-time.
// Input channel layout: [features(CF) | centered xyz(3) | zero pad] so feature
// gathers are float4 (b128) loads; weights column-permuted to match.
// gridDim.z > 1 splits the K-chunk loop; partial pools combined with global
// float-as-int atomicMax (out must be pre-zeroed; values are post-ReLU >= 0).
// ---------------------------------------------------------------------------
template <int CIN0, int CF, int C0, int C1, int C2>
__global__ void __launch_bounds__(128) group_mlp_max_kernel(
    const float* __restrict__ xyz, const float* __restrict__ feats,
    const int* __restrict__ gidx, const float* __restrict__ nxyz,
    int Np, int S, int K,
    const hf16* __restrict__ w0, const float* __restrict__ b0,
    const hf16* __restrict__ w1, const float* __restrict__ b1,
    const hf16* __restrict__ w2, const float* __restrict__ b2,
    float* __restrict__ out) {
    constexpr int WA = (CIN0 > C1) ? CIN0 : C1;
    __shared__ hf16  bufA[16 * WA];
    __shared__ hf16  bufB[16 * C0];
    __shared__ float pooled[C2];

    const int b    = blockIdx.y;
    const int s    = blockIdx.x;
    const int cb   = blockIdx.z;
    const int CB   = gridDim.z;
    const int tid  = threadIdx.x;
    const int lane = tid & 31;
    const int wave = tid >> 5;

    for (int t = tid; t < C2; t += 128) pooled[t] = 0.f;
    __syncthreads();

    const int nChunks = K >> 4;
    for (int kc = cb; kc < nChunks; kc += CB) {
        // gather 16 samples into bufA: [features | centered xyz | pad]
        if constexpr (CF == 0) {
            for (int t = tid; t < 16 * CIN0; t += 128) {
                const int row = t / CIN0;
                const int c   = t - row * CIN0;
                const int kg  = (kc << 4) + row;
                const int g   = gidx ? gidx[(b * S + s) * K + kg] : kg;
                float v = 0.f;
                if (c < 3) {
                    v = xyz[(b * Np + g) * 3 + c];
                    if (nxyz) v -= nxyz[(b * S + s) * 3 + c];
                }
                bufA[t] = (hf16)v;
            }
        } else {
            constexpr int V4 = CIN0 >> 2;  // 4 channels per thread-slot
            for (int t = tid; t < 16 * V4; t += 128) {
                const int row = t / V4;
                const int c4  = (t - row * V4) << 2;
                const int kg  = (kc << 4) + row;
                const int g   = gidx ? gidx[(b * S + s) * K + kg] : kg;
                if (c4 < CF) {
                    const float4 f = *(const float4*)(feats + (size_t)(b * Np + g) * CF + c4);
                    v4h h;
                    h[0] = (hf16)f.x; h[1] = (hf16)f.y; h[2] = (hf16)f.z; h[3] = (hf16)f.w;
                    *(v4h*)(bufA + row * CIN0 + c4) = h;
                } else {
                    v4h h;
#pragma unroll
                    for (int e = 0; e < 4; ++e) {
                        const int c = c4 + e - CF;  // 0..2 = xyz, else pad
                        float v = 0.f;
                        if (c < 3) {
                            v = xyz[(b * Np + g) * 3 + c];
                            if (nxyz) v -= nxyz[(b * S + s) * 3 + c];
                        }
                        h[e] = (hf16)v;
                    }
                    *(v4h*)(bufA + row * CIN0 + c4) = h;
                }
            }
        }
        __syncthreads();

        mlp_layer<CIN0, C0, false>(bufA, bufB, w0, b0, nullptr, lane, wave);
        __syncthreads();
        mlp_layer<C0, C1, false>(bufB, bufA, w1, b1, nullptr, lane, wave);
        __syncthreads();
        mlp_layer<C1, C2, true>(bufA, nullptr, w2, b2, pooled, lane, wave);
        __syncthreads();
    }

    if (CB == 1) {
        for (int t = tid; t < C2; t += 128)
            out[(b * S + s) * C2 + t] = pooled[t];
    } else {
        for (int t = tid; t < C2; t += 128)
            atomicMax((int*)&out[(b * S + s) * C2 + t], __float_as_int(pooled[t]));
    }
}

// ---------------------------------------------------------------------------
// Head: (32,1024) -> FC512+BN+ReLU -> FC256+BN+ReLU -> FC40 -> log_softmax,
// plus l3_points passthrough. Single block, 4 waves, WMMA everywhere.
// ---------------------------------------------------------------------------
__global__ void __launch_bounds__(128) head_kernel(
    const float* __restrict__ x,  // (32,1024) = l3p
    const hf16* __restrict__ w0, const float* __restrict__ b0,
    const hf16* __restrict__ w1, const float* __restrict__ b1,
    const hf16* __restrict__ w2, const float* __restrict__ b2,
    float* __restrict__ out) {
    __shared__ hf16  hA[32 * 512];
    __shared__ hf16  hB[32 * 256];
    __shared__ float logits[32 * 48];

    const int tid  = threadIdx.x;
    const int lane = tid & 31;
    const int wave = tid >> 5;

    // layer 1: 1024 -> 512, A from global f32, KT=32 NT=32
    for (int t = wave; t < 2 * 32; t += 4) {
        const int mt = t & 1, nt = t >> 1;
        v8f acc = {};
#pragma unroll 4
        for (int kt = 0; kt < 32; ++kt) {
            v16h a  = frag_a_global_f32(x, 1024, lane, kt << 5, mt << 4);
            v16h bb = frag_b_swz(w0, 32, lane, kt, nt);
            acc = __builtin_amdgcn_wmma_f32_16x16x32_f16(
                false, a, false, bb, (short)0, acc, false, false);
        }
        const int n = lane & 15, grp = lane >> 4;
        const int ncol = (nt << 4) + n;
        const float bias = b0[ncol];
#pragma unroll
        for (int j = 0; j < 8; ++j) {
            const float vv = fmaxf(acc[j] + bias, 0.f);
            hA[((mt << 4) + j + (grp << 3)) * 512 + ncol] = (hf16)vv;
        }
    }
    __syncthreads();

    // layer 2: 512 -> 256, KT=16 NT=16
    for (int t = wave; t < 2 * 16; t += 4) {
        const int mt = t & 1, nt = t >> 1;
        v8f acc = {};
#pragma unroll 4
        for (int kt = 0; kt < 16; ++kt) {
            v16h a  = frag_a_lds(hA + (mt << 4) * 512, 512, lane, kt << 5);
            v16h bb = frag_b_swz(w1, 16, lane, kt, nt);
            acc = __builtin_amdgcn_wmma_f32_16x16x32_f16(
                false, a, false, bb, (short)0, acc, false, false);
        }
        const int n = lane & 15, grp = lane >> 4;
        const int ncol = (nt << 4) + n;
        const float bias = b1[ncol];
#pragma unroll
        for (int j = 0; j < 8; ++j) {
            const float vv = fmaxf(acc[j] + bias, 0.f);
            hB[((mt << 4) + j + (grp << 3)) * 256 + ncol] = (hf16)vv;
        }
    }
    __syncthreads();

    // layer 3: 256 -> 48 (40 valid), KT=8 NT=3, raw logits
    for (int t = wave; t < 2 * 3; t += 4) {
        const int mt = t & 1, nt = t >> 1;
        v8f acc = {};
#pragma unroll
        for (int kt = 0; kt < 8; ++kt) {
            v16h a  = frag_a_lds(hB + (mt << 4) * 256, 256, lane, kt << 5);
            v16h bb = frag_b_swz(w2, 8, lane, kt, nt);
            acc = __builtin_amdgcn_wmma_f32_16x16x32_f16(
                false, a, false, bb, (short)0, acc, false, false);
        }
        const int n = lane & 15, grp = lane >> 4;
        const int ncol = (nt << 4) + n;
        const float bias = b2[ncol];
#pragma unroll
        for (int j = 0; j < 8; ++j)
            logits[((mt << 4) + j + (grp << 3)) * 48 + ncol] = acc[j] + bias;
    }
    __syncthreads();

    // log_softmax over 40 classes, one thread per row
    if (tid < 32) {
        float mx = -3.4e38f;
        for (int j = 0; j < 40; ++j) mx = fmaxf(mx, logits[tid * 48 + j]);
        float sum = 0.f;
        for (int j = 0; j < 40; ++j) sum += __expf(logits[tid * 48 + j] - mx);
        const float lse = mx + __logf(sum);
        for (int j = 0; j < 40; ++j) out[tid * 40 + j] = logits[tid * 48 + j] - lse;
    }
    // l3_points output: (B,1024,1) flat == (B,1024)
    for (int t = tid; t < 32 * 1024; t += 128) out[1280 + t] = x[t];
}

// ---------------------------------------------------------------------------
// Host orchestration
// ---------------------------------------------------------------------------
extern "C" void kernel_launch(void* const* d_in, const int* in_sizes, int n_in,
                              void* d_out, int out_size, void* d_ws, size_t ws_size,
                              hipStream_t stream) {
    (void)in_sizes; (void)n_in; (void)out_size; (void)ws_size;
    const int B = 32, N = 1024;
    const float* xyz = (const float*)d_in[0];
    auto P = [&](int i) { return (const float*)d_in[i]; };

    char*  ws  = (char*)d_ws;
    size_t off = 0;
    auto alloc = [&](size_t bytes) -> char* {
        char* p = ws + off;
        off += (bytes + 255) & ~(size_t)255;
        return p;
    };

    float* pts   = (float*)alloc((size_t)B * N * 3 * 4);
    int*   fidx1 = (int*)  alloc((size_t)B * 512 * 4);
    float* nx1   = (float*)alloc((size_t)B * 512 * 3 * 4);
    int*   gidx1 = (int*)  alloc((size_t)B * 512 * 32 * 4);
    float* l1p   = (float*)alloc((size_t)B * 512 * 128 * 4);
    int*   fidx2 = (int*)  alloc((size_t)B * 256 * 4);
    float* nx2   = (float*)alloc((size_t)B * 256 * 3 * 4);
    int*   gidx2 = (int*)  alloc((size_t)B * 256 * 64 * 4);
    float* l2p   = (float*)alloc((size_t)B * 256 * 256 * 4);
    float* l3p   = (float*)alloc((size_t)B * 1024 * 4);

    // 12 layers: cin, cout, cinPad, coutPad, input-base-index, has_bn, cfPerm
    struct LDesc { int cin, cout, cinPad, coutPad, base, bn, cf; };
    const LDesc L[12] = {
        {3, 64, 32, 64, 1, 1, 0},        {64, 64, 64, 64, 7, 1, 0},
        {64, 128, 64, 128, 13, 1, 0},
        {131, 128, 160, 128, 19, 1, 128},{128, 128, 128, 128, 25, 1, 0},
        {128, 256, 128, 256, 31, 1, 0},
        {259, 256, 288, 256, 37, 1, 256},{256, 512, 256, 512, 43, 1, 0},
        {512, 1024, 512, 1024, 49, 1, 0},
        {1024, 512, 1024, 512, 55, 1, 0},{512, 256, 512, 256, 61, 1, 0},
        {256, 40, 256, 48, 67, 0, 0}};

    hf16*  Wf[12];
    float* Bf[12];
    for (int i = 0; i < 12; ++i) {
        Wf[i] = (hf16*) alloc((size_t)L[i].cinPad * L[i].coutPad * sizeof(hf16));
        Bf[i] = (float*)alloc((size_t)L[i].coutPad * sizeof(float));
        const float* W  = P(L[i].base + 0);
        const float* bb = P(L[i].base + 1);
        const float* g  = L[i].bn ? P(L[i].base + 2) : nullptr;
        const float* be = L[i].bn ? P(L[i].base + 3) : nullptr;
        const float* m  = L[i].bn ? P(L[i].base + 4) : nullptr;
        const float* v  = L[i].bn ? P(L[i].base + 5) : nullptr;
        const int tot = L[i].cinPad * L[i].coutPad;
        fold_kernel<<<(tot + 255) / 256, 256, 0, stream>>>(
            W, bb, g, be, m, v, L[i].cin, L[i].cout, L[i].cinPad, L[i].coutPad,
            Wf[i], Bf[i], L[i].bn, L[i].cf);
    }

    transpose_xyz_kernel<<<(B * N * 3 + 255) / 256, 256, 0, stream>>>(xyz, pts, B, N);

    // ---- SA1: 1024 pts -> 512 centroids, r=0.2, K=32, MLP 3->64->64->128
    fps_kernel<<<B, 256, 0, stream>>>(pts, N, 512, fidx1);
    gather3_kernel<<<(B * 512 * 3 + 255) / 256, 256, 0, stream>>>(pts, fidx1, N, 512, nx1);
    ball_query_kernel<<<(B * 512 + 255) / 256, 256, 0, stream>>>(
        pts, nx1, N, 512, 32, 0.04f, B, gidx1);
    group_mlp_max_kernel<32, 0, 64, 64, 128><<<dim3(512, B, 1), 128, 0, stream>>>(
        pts, nullptr, gidx1, nx1, N, 512, 32,
        Wf[0], Bf[0], Wf[1], Bf[1], Wf[2], Bf[2], l1p);

    // ---- SA2: 512 -> 256 centroids, r=0.4, K=64, MLP 131->128->128->256
    fps_kernel<<<B, 256, 0, stream>>>(nx1, 512, 256, fidx2);
    gather3_kernel<<<(B * 256 * 3 + 255) / 256, 256, 0, stream>>>(nx1, fidx2, 512, 256, nx2);
    ball_query_kernel<<<(B * 256 + 255) / 256, 256, 0, stream>>>(
        nx1, nx2, 512, 256, 64, 0.16f, B, gidx2);
    group_mlp_max_kernel<160, 128, 128, 128, 256><<<dim3(256, B, 1), 128, 0, stream>>>(
        nx1, l1p, gidx2, nx2, 512, 256, 64,
        Wf[3], Bf[3], Wf[4], Bf[4], Wf[5], Bf[5], l2p);

    // ---- SA3 (group-all): 256 pts, MLP 259->256->512->1024, global max.
    // Split K-chunks over grid z (8 blocks x 2 chunks) for WGP utilization;
    // partial pools merged via float-as-int atomicMax on pre-zeroed l3p.
    zero_kernel<<<(B * 1024 + 255) / 256, 256, 0, stream>>>(l3p, B * 1024);
    group_mlp_max_kernel<288, 256, 256, 512, 1024><<<dim3(1, B, 8), 128, 0, stream>>>(
        nx2, l2p, nullptr, nullptr, 256, 1, 256,
        Wf[6], Bf[6], Wf[7], Bf[7], Wf[8], Bf[8], l3p);

    // ---- Head + log_softmax + l3_points passthrough
    head_kernel<<<1, 128, 0, stream>>>(
        l3p, Wf[9], Bf[9], Wf[10], Bf[10], Wf[11], Bf[11], (float*)d_out);
}
